// INTQuantBase_83519934038116
// MI455X (gfx1250) — 8-bit matrix operand order repaired and checked by reading, compiled, not run on hardware
//
#include <hip/hip_runtime.h>


typedef __attribute__((ext_vector_type(8)))  int   v8i;
typedef __attribute__((ext_vector_type(4)))  int   v4i;
typedef __attribute__((ext_vector_type(2)))  int   v2i;
typedef __attribute__((ext_vector_type(4)))  float v4f;

#define B_    32
#define CIN   16
#define COUT  16
#define H_    256
#define W_    256
#define HP    (H_ + 2)
#define WP    (W_ + 2)
#define NPIX  (B_ * HP * WP)
#define NELEM (B_ * CIN * H_ * W_)
#define ROWS  4

#define WS_WPACK_OFF 64
#define WS_XQ_OFF    4096

__global__ void init_kernel(unsigned* __restrict__ maxbits) { *maxbits = 0u; }

__global__ __launch_bounds__(256)
void absmax_kernel(const float4* __restrict__ x4, unsigned* __restrict__ maxbits, int n4) {
    __shared__ unsigned sm[256];
    unsigned m = 0u;
    for (int i = blockIdx.x * 256 + threadIdx.x; i < n4; i += gridDim.x * 256) {
        float4 v = x4[i];
        unsigned a = __float_as_uint(v.x) & 0x7fffffffu; m = a > m ? a : m;
        a = __float_as_uint(v.y) & 0x7fffffffu;          m = a > m ? a : m;
        a = __float_as_uint(v.z) & 0x7fffffffu;          m = a > m ? a : m;
        a = __float_as_uint(v.w) & 0x7fffffffu;          m = a > m ? a : m;
    }
    sm[threadIdx.x] = m;
    __syncthreads();
    for (int s = 128; s > 0; s >>= 1) {
        if ((int)threadIdx.x < s) {
            unsigned o = sm[threadIdx.x + s];
            if (o > sm[threadIdx.x]) sm[threadIdx.x] = o;
        }
        __syncthreads();
    }
    if (threadIdx.x == 0) atomicMax(maxbits, sm[0]);
}

__global__ void packw_kernel(const float* __restrict__ w, int* __restrict__ wpack) {
    int t = threadIdx.x;
    if (t >= 96) return;
    int ky = t / 32, l = t % 32, n = l & 15;
    v8i frag = {0, 0, 0, 0, 0, 0, 0, 0};
#pragma unroll
    for (int r = 0; r < 8; ++r) {
        int word = 0;
#pragma unroll
        for (int j = 0; j < 4; ++j) {
            int k = (r >> 1) * 16 + (l >> 4) * 8 + (r & 1) * 4 + j;
            int byte = 0;
            if (k < 48) {
                int kx = k >> 4, cin = k & 15;
                float wv = w[((n * CIN + cin) * 3 + ky) * 3 + kx];
                byte = ((int)rintf(fminf(fmaxf(wv, -127.0f), 127.0f))) & 0xFF;
            }
            word |= byte << (8 * j);
        }
        frag[r] = word;
    }
    *(v8i*)(wpack + t * 8) = frag;
}

__global__ __launch_bounds__(256)
void quant_pad_kernel(const float* __restrict__ x, const unsigned* __restrict__ maxbits,
                      signed char* __restrict__ xq) {
    int p = blockIdx.x * 256 + threadIdx.x;
    if (p >= NPIX) return;
    int wp = p % WP;
    int t  = p / WP;
    int hp = t % HP;
    int b  = t / HP;
    v4i pack = {0, 0, 0, 0};
    if (hp >= 1 && hp <= H_ && wp >= 1 && wp <= W_) {
        const float inv = 128.0f / __uint_as_float(*maxbits);
        const float* xb = x + (long)b * (CIN * H_ * W_) + (hp - 1) * W_ + (wp - 1);
        int v[16];
#pragma unroll
        for (int c = 0; c < 16; ++c) {
            float q = rintf(fminf(fmaxf(xb[(long)c * (H_ * W_)] * inv, -127.0f), 127.0f));
            v[c] = ((int)q) & 0xFF;
        }
        pack.x = v[0]  | (v[1]  << 8) | (v[2]  << 16) | (v[3]  << 24);
        pack.y = v[4]  | (v[5]  << 8) | (v[6]  << 16) | (v[7]  << 24);
        pack.z = v[8]  | (v[9]  << 8) | (v[10] << 16) | (v[11] << 24);
        pack.w = v[12] | (v[13] << 8) | (v[14] << 16) | (v[15] << 24);
    }
    *(v4i*)(xq + (long)p * 16) = pack;
}

__global__ __launch_bounds__(256)
void conv_wmma_kernel(const signed char* __restrict__ xq,
                      const int* __restrict__ wpack,
                      const unsigned* __restrict__ maxbits,
                      const float* __restrict__ scale_w,
                      const float* __restrict__ bias,
                      float* __restrict__ out) {
    const int lane = threadIdx.x & 31;
    const int wave = threadIdx.x >> 5;
    const int tile = blockIdx.x * 8 + wave;
    const int wt = tile & 15;
    const int hb = (tile >> 4) & 63;
    const int b  = tile >> 10;

    const int m    = lane & 15;
    const int half = lane >> 4;
    const int h0   = hb * ROWS;

    v8i bf[3];
#pragma unroll
    for (int ky = 0; ky < 3; ++ky)
        bf[ky] = *(const v8i*)(wpack + (ky * 32 + lane) * 8);

    const char* abase = (const char*)xq
        + ((((long)b * HP + h0) * WP + wt * 16 + m) * 16 + half * 8);

    v8i acc[ROWS];
#pragma unroll
    for (int r = 0; r < ROWS; ++r) acc[r] = (v8i){0, 0, 0, 0, 0, 0, 0, 0};

#pragma unroll
    for (int ir = 0; ir < ROWS + 2; ++ir) {
        const char* p = abase + (long)ir * (WP * 16);
        v2i t0 = *(const v2i*)(p + 0);
        v2i t1 = *(const v2i*)(p + 16);
        v2i t2 = *(const v2i*)(p + 32);
        v8i a = {t0.x, t0.y, t1.x, t1.y, t2.x, t2.y, 0, 0};
#pragma unroll
        for (int ky = 0; ky < 3; ++ky) {
            const int r = ir - ky;
            if (r >= 0 && r < ROWS) {
                acc[r] = __builtin_amdgcn_wmma_i32_16x16x64_iu8(
                    true, a, true, bf[ky], acc[r], false, false);
            }
        }
    }

    const int   n    = lane & 15;
    const float s_in = __uint_as_float(*maxbits) * (1.0f / 128.0f);
    const float f    = s_in * scale_w[n];
    const float bb   = bias[n];
    const int   w0   = wt * 16 + half * 8;

#pragma unroll
    for (int r = 0; r < ROWS; ++r) {
        float* o = out + (((long)b * COUT + n) * H_ + (h0 + r)) * W_ + w0;
        v4f o0 = {(float)acc[r][0] * f + bb, (float)acc[r][1] * f + bb,
                  (float)acc[r][2] * f + bb, (float)acc[r][3] * f + bb};
        v4f o1 = {(float)acc[r][4] * f + bb, (float)acc[r][5] * f + bb,
                  (float)acc[r][6] * f + bb, (float)acc[r][7] * f + bb};
        *(v4f*)(o)     = o0;
        *(v4f*)(o + 4) = o1;
    }
}

extern "C" void kernel_launch(void* const* d_in, const int* in_sizes, int n_in,
                              void* d_out, int out_size, void* d_ws, size_t ws_size,
                              hipStream_t stream) {
    const float* x    = (const float*)d_in[0];
    const float* w8   = (const float*)d_in[1];
    const float* sw   = (const float*)d_in[2];
    const float* bias = (const float*)d_in[3];
    float* out = (float*)d_out;

    char* ws = (char*)d_ws;
    unsigned*    maxbits = (unsigned*)ws;
    int*         wpack   = (int*)(ws + WS_WPACK_OFF);
    signed char* xq      = (signed char*)(ws + WS_XQ_OFF);

    init_kernel<<<1, 1, 0, stream>>>(maxbits);

    const int n4 = NELEM / 4;
    absmax_kernel<<<4096, 256, 0, stream>>>((const float4*)x, maxbits, n4);

    packw_kernel<<<1, 128, 0, stream>>>(w8, wpack);

    const int qblocks = (NPIX + 255) / 256;
    quant_pad_kernel<<<qblocks, 256, 0, stream>>>(x, maxbits, xq);

    conv_wmma_kernel<<<4096, 256, 0, stream>>>(xq, wpack, maxbits, sw, bias, out);
}
